// SelfAttentionNarrow_21852793602052
// MI455X (gfx1250) — matrix operand order repaired and checked by reading, not compiled, not run on hardware
//
#include <hip/hip_runtime.h>
#include <stdint.h>

#define TSEQ  2048
#define BATCH 4
#define NHEAD 16
#define EMBD  1024
#define HS    64
#define NBH   64   // BATCH*NHEAD

typedef __attribute__((ext_vector_type(16))) _Float16 v16h;
typedef __attribute__((ext_vector_type(8)))  float    v8f;

union F16Frag { v16h v; uint32_t u[8]; _Float16 h[16]; };

__device__ inline uint32_t pack_f16x2(float a, float b) {
  union { _Float16 h[2]; uint32_t u; } p;
  p.h[0] = (_Float16)a; p.h[1] = (_Float16)b;
  return p.u;
}

__device__ inline void async_b128(uint32_t lds_off, const void* gaddr) {
  asm volatile("global_load_async_to_lds_b128 %0, %1, off"
               :: "v"(lds_off), "v"(gaddr) : "memory");
}
__device__ inline void wait_async0() {
  asm volatile("s_wait_asynccnt 0" ::: "memory");
}

// ---------------------------------------------------------------- kernel 0a
__global__ __launch_bounds__(256) void wu_to_f16(const float* __restrict__ Wu,
                                                 _Float16* __restrict__ Wuh) {
  int i = (blockIdx.x * 256 + threadIdx.x) * 4;
  float4 f = *(const float4*)(Wu + i);
  Wuh[i + 0] = (_Float16)f.x;
  Wuh[i + 1] = (_Float16)f.y;
  Wuh[i + 2] = (_Float16)f.z;
  Wuh[i + 3] = (_Float16)f.w;
}

// ---------------------------------------------------------------- kernel 0b
// Pack Wq/Wk/Wv (each 64x64 f32) into one contiguous f16 buffer:
// Wh[0]=Wq, Wh[1]=Wk, Wh[2]=Wv (row-major (o,s) preserved).
__global__ __launch_bounds__(256) void wsmall_to_f16(
    const float* __restrict__ Wq, const float* __restrict__ Wk,
    const float* __restrict__ Wv, _Float16* __restrict__ Wh) {
  int i = blockIdx.x * 256 + threadIdx.x;          // 0 .. 3*4096-1
  int w = i >> 12, idx = i & 4095;
  const float* src = (w == 0) ? Wq : (w == 1) ? Wk : Wv;
  Wh[i] = (_Float16)src[idx];
}

// ---------------------------------------------------------------- kernel 1
// Q/K/V projection. One wave computes the 16x64 output tile for all three
// matrices for one (b,t): the 16 rows are the 16 heads (each a 64-vector).
// Weights async-staged once into LDS (24KB), shared by the 4 waves.
// Q pre-scaled by 1/32 (= 1/sqrt(EMB), exact). V stored transposed (n,s,t).
__global__ __launch_bounds__(128) void qkv_kernel(
    const float* __restrict__ x, const _Float16* __restrict__ Wh,
    _Float16* __restrict__ Qh, _Float16* __restrict__ Kh,
    _Float16* __restrict__ Vt) {
  __shared__ __align__(16) _Float16 sW[3 * 64 * 64];   // 24KB: [w][o][s]
  const int tid = threadIdx.x;
  const int lane = tid & 31, wave = tid >> 5;
  const int hh = lane >> 4, ln = lane & 15;
  const int tile = blockIdx.x * 4 + wave;      // == b*TSEQ + t
  const int b = tile / TSEQ, t = tile % TSEQ;

  // async-stage all three weight matrices (24KB contiguous, 1536 x 16B)
  {
    const uint32_t ldsW = (uint32_t)(uintptr_t)sW;
#pragma unroll
    for (int p = 0; p < 12; ++p) {
      int c = p * 128 + tid;
      async_b128(ldsW + c * 16, (const char*)Wh + c * 16);
    }
  }

  // A fragments (16x32 f16): row m = head = ln, k = s  (overlaps async)
  F16Frag af[2];
  const float* xrow = x + ((size_t)tile * NHEAD + ln) * HS;
#pragma unroll
  for (int c = 0; c < 2; ++c)
#pragma unroll
    for (int v = 0; v < 8; ++v) {
      int k = 32 * c + hh * 16 + 2 * v;
      af[c].u[v] = pack_f16x2(xrow[k], xrow[k + 1]);
    }

  wait_async0();
  __syncthreads();

#pragma unroll
  for (int w = 0; w < 3; ++w) {
#pragma unroll
    for (int j = 0; j < 4; ++j) {
      v8f acc = {};
#pragma unroll
      for (int c = 0; c < 2; ++c) {
        F16Frag bf;                                    // B = W^T (32x16)
        const _Float16* wp =
            sW + w * 4096 + (j * 16 + ln) * HS + 32 * c + hh * 16;
#pragma unroll
        for (int v = 0; v < 8; ++v) bf.u[v] = *(const uint32_t*)(wp + 2 * v);
        acc = __builtin_amdgcn_wmma_f32_16x16x32_f16(
            false, af[c].v, false, bf.v, (short)0, acc, false, false);
      }
#pragma unroll
      for (int g = 0; g < 8; ++g) {
        int head = g + 8 * hh;                         // C row
        int s = j * 16 + ln;                           // C col
        int nh = b * NHEAD + head;
        float val = acc[g];
        if (w == 0)
          Qh[((size_t)nh * TSEQ + t) * HS + s] = (_Float16)(val * 0.03125f);
        else if (w == 1)
          Kh[((size_t)nh * TSEQ + t) * HS + s] = (_Float16)val;
        else
          Vt[((size_t)nh * HS + s) * TSEQ + t] = (_Float16)val;
      }
    }
  }
}

// ---------------------------------------------------------------- kernel 2
// Flash attention. WG = (head n, 64 q rows); wave owns 16 q rows.
// K (32x64) and V^T (64x32) blocks staged in LDS via async-to-LDS, shared
// by all 4 waves. Reproduces reference masking bug: only n<4 masked.
__global__ __launch_bounds__(128) void attn_kernel(
    const _Float16* __restrict__ Qh, const _Float16* __restrict__ Kh,
    const _Float16* __restrict__ Vt, const long long* __restrict__ pad,
    _Float16* __restrict__ AO) {
  __shared__ __align__(16) _Float16 sK[32 * 64];      // [key][s]
  __shared__ __align__(16) _Float16 sV[64 * 32];      // [s][key]
  __shared__ __align__(16) _Float16 sP[4][16 * 32];   // per-wave P transpose

  const int tid = threadIdx.x;
  const int lane = tid & 31, wave = tid >> 5;
  const int hh = lane >> 4, ln = lane & 15;
  const int n = blockIdx.y;                            // head 0..63
  const int qbase = blockIdx.x * 64 + wave * 16;
  int thresh = TSEQ;
  if (n < BATCH) thresh = TSEQ - (int)pad[n];

  // Q A-fragments (held for whole K loop)
  F16Frag qf[2];
  const _Float16* qrow = Qh + ((size_t)n * TSEQ + qbase + ln) * HS;
#pragma unroll
  for (int c = 0; c < 2; ++c)
#pragma unroll
    for (int v = 0; v < 8; ++v) {
      int k = 32 * c + hh * 16 + 2 * v;
      qf[c].u[v] = *(const uint32_t*)(qrow + k);
    }

  v8f o[4] = {{}, {}, {}, {}};
  float mrow[8], lrow[8];
#pragma unroll
  for (int g = 0; g < 8; ++g) { mrow[g] = -3.0e38f; lrow[g] = 0.0f; }

  const uint32_t ldsK = (uint32_t)(uintptr_t)sK;
  const uint32_t ldsV = (uint32_t)(uintptr_t)sV;

  for (int kb = 0; kb < TSEQ / 32; ++kb) {
    __syncthreads();  // previous block fully consumed
    // ---- async stage: K block is 4KB contiguous; V^T block is 64 rows x 64B
    {
      const char* kg = (const char*)(Kh + ((size_t)n * TSEQ + kb * 32) * HS);
#pragma unroll
      for (int p = 0; p < 2; ++p) {
        int c = p * 128 + tid;                         // 256 x 16B chunks
        async_b128(ldsK + c * 16, kg + c * 16);
      }
#pragma unroll
      for (int p = 0; p < 2; ++p) {
        int c = p * 128 + tid;                         // s = c/4, 16B sub
        int srow = c >> 2, off = (c & 3) * 16;
        const void* g =
            (const char*)(Vt + ((size_t)n * HS + srow) * TSEQ + kb * 32) + off;
        async_b128(ldsV + srow * 64 + off, g);
      }
    }
    wait_async0();
    __syncthreads();  // LDS visible to all waves

    // ---- scores: two 16-key tiles, K-dim = s (64) -> 2 WMMAs each
    v8f sc[2];
#pragma unroll
    for (int jt = 0; jt < 2; ++jt) {
      v8f a = {};
#pragma unroll
      for (int c = 0; c < 2; ++c) {
        F16Frag kf;
        const _Float16* kp = sK + (jt * 16 + ln) * HS + 32 * c + hh * 16;
#pragma unroll
        for (int v = 0; v < 8; ++v) kf.u[v] = *(const uint32_t*)(kp + 2 * v);
        a = __builtin_amdgcn_wmma_f32_16x16x32_f16(
            false, qf[c].v, false, kf.v, (short)0, a, false, false);
      }
      sc[jt] = a;
    }

    // ---- mask (reference bug: only heads n<4 of b=0)
    if (thresh < TSEQ) {
#pragma unroll
      for (int jt = 0; jt < 2; ++jt) {
        int kc = kb * 32 + jt * 16 + ln;
#pragma unroll
        for (int g = 0; g < 8; ++g) {
          int qr = qbase + g + 8 * hh;
          if (qr >= thresh && kc >= thresh) sc[jt][g] = -3.0e38f;
        }
      }
    }

    // ---- online softmax; rows live in 16-lane groups -> xor reductions
#pragma unroll
    for (int g = 0; g < 8; ++g) {
      float m = fmaxf(sc[0][g], sc[1][g]);
      m = fmaxf(m, __shfl_xor(m, 1));
      m = fmaxf(m, __shfl_xor(m, 2));
      m = fmaxf(m, __shfl_xor(m, 4));
      m = fmaxf(m, __shfl_xor(m, 8));
      float mnew = fmaxf(mrow[g], m);
      float alpha = __expf(mrow[g] - mnew);
      float p0 = __expf(sc[0][g] - mnew);
      float p1 = __expf(sc[1][g] - mnew);
      sc[0][g] = p0; sc[1][g] = p1;
      float ps = p0 + p1;
      ps += __shfl_xor(ps, 1);
      ps += __shfl_xor(ps, 2);
      ps += __shfl_xor(ps, 4);
      ps += __shfl_xor(ps, 8);
      lrow[g] = lrow[g] * alpha + ps;
      mrow[g] = mnew;
#pragma unroll
      for (int j = 0; j < 4; ++j) o[j][g] *= alpha;
    }

    // ---- P: C-layout -> A-layout via per-wave LDS bounce (in-order DS)
    _Float16* pw = sP[wave];
#pragma unroll
    for (int jt = 0; jt < 2; ++jt)
#pragma unroll
      for (int g = 0; g < 8; ++g)
        pw[(g + 8 * hh) * 32 + jt * 16 + ln] = (_Float16)sc[jt][g];
    F16Frag pf;
    {
      const _Float16* pr = pw + ln * 32;
#pragma unroll
      for (int v = 0; v < 8; ++v) {
        int k = hh * 16 + 2 * v;
        pf.u[v] = *(const uint32_t*)(pr + k);
      }
    }

    // ---- O += P(16x32) * V(32x64): 4 WMMAs
#pragma unroll
    for (int j = 0; j < 4; ++j) {
      F16Frag vf;
      const _Float16* vp = sV + (j * 16 + ln) * 32 + hh * 16;
#pragma unroll
      for (int v = 0; v < 8; ++v) vf.u[v] = *(const uint32_t*)(vp + 2 * v);
      o[j] = __builtin_amdgcn_wmma_f32_16x16x32_f16(
          false, pf.v, false, vf.v, (short)0, o[j], false, false);
    }
  }

  // ---- normalize + store to (b, t, h*S + s) f16
  const int b = n >> 4, h = n & 15;
#pragma unroll
  for (int g = 0; g < 8; ++g) {
    float inv = 1.0f / lrow[g];
    int t = qbase + g + 8 * hh;
    _Float16* op = AO + ((size_t)(b * TSEQ + t)) * EMBD + h * HS;
#pragma unroll
    for (int j = 0; j < 4; ++j) op[j * 16 + ln] = (_Float16)(o[j][g] * inv);
  }
}

// ---------------------------------------------------------------- kernel 3
// out = AO(8192x1024) @ Wu^T + bu, f16 WMMA, f32 out.
// WG = 128 rows x 64 cols; B slab (64n x 128k, 16KB) async-staged in LDS and
// shared by all 4 waves; each wave does M=32 rows (2x B-fragment reuse).
__global__ __launch_bounds__(128) void proj_kernel(
    const _Float16* __restrict__ AO, const _Float16* __restrict__ Wuh,
    const float* __restrict__ bu, float* __restrict__ out) {
  __shared__ __align__(16) _Float16 sB[64 * 128];     // [n][k] 16KB
  const int tid = threadIdx.x;
  const int lane = tid & 31, wave = tid >> 5;
  const int hh = lane >> 4, ln = lane & 15;
  const int mbase = blockIdx.x * 128 + wave * 32;
  const int ncol = blockIdx.y * 64;
  const uint32_t ldsB = (uint32_t)(uintptr_t)sB;

  v8f acc[8] = {{}, {}, {}, {}, {}, {}, {}, {}};   // [rowset 0..1][j 0..3]

  for (int ks = 0; ks < EMBD / 128; ++ks) {
    __syncthreads();  // previous slab consumed
    // stage B slab: 128 rows x 128B (256B LDS row stride) = 1024 x 16B chunks, 8 per thread
#pragma unroll
    for (int p = 0; p < 8; ++p) {
      int c = p * 128 + tid;
      int nrow = c >> 3, off = (c & 7) * 16;
      const void* g =
          (const char*)(Wuh + (size_t)(ncol + nrow) * EMBD + ks * 128) + off;
      async_b128(ldsB + nrow * 256 + off, g);
    }
    wait_async0();
    __syncthreads();

#pragma unroll
    for (int s4 = 0; s4 < 4; ++s4) {                 // 32-k sub-chunks
      int kc = ks * 128 + s4 * 32;
      F16Frag af[2];
#pragma unroll
      for (int r = 0; r < 2; ++r) {
        const _Float16* ap = AO + (size_t)(mbase + r * 16 + ln) * EMBD + kc;
        __builtin_prefetch(ap + 128, 0, 0);          // next slab's A chunk
#pragma unroll
        for (int v = 0; v < 8; ++v) {
          int k = hh * 16 + 2 * v;
          af[r].u[v] = *(const uint32_t*)(ap + k);
        }
      }
#pragma unroll
      for (int j = 0; j < 4; ++j) {
        F16Frag bf;
        const _Float16* bp = sB + (j * 16 + ln) * 128 + s4 * 32 + hh * 16;
#pragma unroll
        for (int v = 0; v < 8; ++v) bf.u[v] = *(const uint32_t*)(bp + 2 * v);
        acc[j] = __builtin_amdgcn_wmma_f32_16x16x32_f16(
            false, af[0].v, false, bf.v, (short)0, acc[j], false, false);
        acc[4 + j] = __builtin_amdgcn_wmma_f32_16x16x32_f16(
            false, af[1].v, false, bf.v, (short)0, acc[4 + j], false, false);
      }
    }
  }

#pragma unroll
  for (int j = 0; j < 4; ++j) {
    float bias = bu[ncol + j * 16 + ln];
#pragma unroll
    for (int r = 0; r < 2; ++r)
#pragma unroll
      for (int g = 0; g < 8; ++g) {
        int row = mbase + r * 16 + g + 8 * hh;
        out[(size_t)row * EMBD + ncol + j * 16 + ln] = acc[r * 4 + j][g] + bias;
      }
  }
}

// ---------------------------------------------------------------- launch
extern "C" void kernel_launch(void* const* d_in, const int* in_sizes, int n_in,
                              void* d_out, int out_size, void* d_ws,
                              size_t ws_size, hipStream_t stream) {
  const float* x  = (const float*)d_in[0];
  const float* Wk = (const float*)d_in[1];
  const float* Wq = (const float*)d_in[2];
  const float* Wv = (const float*)d_in[3];
  const float* Wu = (const float*)d_in[4];
  const float* bu = (const float*)d_in[5];
  const long long* pad = (const long long*)d_in[6];
  float* out = (float*)d_out;

  char* ws = (char*)d_ws;
  const size_t MB = 1024 * 1024;
  _Float16* Qh  = (_Float16*)(ws + 0 * MB);    // 16 MB: (n, t, s)
  _Float16* Kh  = (_Float16*)(ws + 16 * MB);   // 16 MB: (n, t, s)
  _Float16* Vt  = (_Float16*)(ws + 32 * MB);   // 16 MB: (n, s, t)
  _Float16* AO  = (_Float16*)(ws + 48 * MB);   // 16 MB: (b, t, emb)
  _Float16* Wuh = (_Float16*)(ws + 64 * MB);   //  2 MB
  _Float16* Wh  = (_Float16*)(ws + 66 * MB);   // 24 KB: Wq|Wk|Wv f16

  wu_to_f16<<<1024, 256, 0, stream>>>(Wu, Wuh);
  wsmall_to_f16<<<48, 256, 0, stream>>>(Wq, Wk, Wv, Wh);
  qkv_kernel<<<2048, 128, 0, stream>>>(x, Wh, Qh, Kh, Vt);
  attn_kernel<<<dim3(TSEQ / 64, NBH), 128, 0, stream>>>(Qh, Kh, Vt, pad, AO);
  proj_kernel<<<dim3(EMBD * 8 / 128 / 8, EMBD / 64), 128, 0, stream>>>(
      AO, Wuh, bu, out);
}
